// FraudDetectionHybrid_65481071408813
// MI455X (gfx1250) — matrix operand order repaired and checked by reading, not compiled, not run on hardware
//
#include <hip/hip_runtime.h>
#include <hip/hip_bf16.h>

// ---------------------------------------------------------------------------
// Fused fraud-detection MLP for gfx1250 (MI455X), wave32 + WMMA f16.
// ---------------------------------------------------------------------------

typedef _Float16 v16h  __attribute__((ext_vector_type(16)));
typedef _Float16 half8 __attribute__((ext_vector_type(8)));
typedef float    v8f   __attribute__((ext_vector_type(8)));
typedef float    fx4   __attribute__((ext_vector_type(4)));

#define INPUT_DIM      512
#define H1_DIM         128
#define H2_DIM         64
#define LAT_DIM        32
#define ROWS_PER_BLOCK 128
#define BLOCK_THREADS  256

// Padded LDS strides (elements) to avoid bank conflicts.
#define LDW1  520   // 512 + 8 halves
#define LDH1  136   // 128 + 8
#define LDW2  136
#define LDW3  72    // 64 + 8
#define LDH2  72
#define LDLAT 33    // 32 + 1 floats

// LDS byte offsets (all 16B aligned)
#define OFF_W1   0
#define OFF_H1   (OFF_W1  + H1_DIM        * LDW1 * 2)   // 133120
#define OFF_W2   (OFF_H1  + ROWS_PER_BLOCK* LDH1 * 2)   // 167936
#define OFF_W3   (OFF_W2  + H2_DIM        * LDW2 * 2)   // 185344
#define OFF_H2   (OFF_W3  + LAT_DIM       * LDW3 * 2)   // 189952
#define OFF_LAT  (OFF_H2  + ROWS_PER_BLOCK* LDH2 * 2)   // 208384
#define OFF_FCL  (OFF_LAT + ROWS_PER_BLOCK* LDLAT * 4)  // 225280
#define SMEM_BYTES (OFF_FCL + 16)                        // 225296 < 320KB

// ---- fragment loaders (per CDNA5 ISA 7.12.2 wave32 VGPR layouts) ----------

// 16-bit A 16x32: lane(0-15)=M, elems 0..7 -> K = 8*lh + 0..7,
//                 elems 8..15 -> K = 16 + 8*lh + 0..7   (lh = lane>=16)
__device__ __forceinline__ v16h a_frag_from_global(const float* p) {
    fx4 x0 = *(const fx4*)(p + 0);
    fx4 x1 = *(const fx4*)(p + 4);
    fx4 x2 = *(const fx4*)(p + 16);
    fx4 x3 = *(const fx4*)(p + 20);
    v16h r;
#pragma unroll
    for (int e = 0; e < 4; ++e) {
        r[e]      = (_Float16)x0[e];
        r[4 + e]  = (_Float16)x1[e];
        r[8 + e]  = (_Float16)x2[e];
        r[12 + e] = (_Float16)x3[e];
    }
    return r;
}

__device__ __forceinline__ v16h a_frag_from_lds(const _Float16* p) {
    half8 lo = *(const half8*)(p);
    half8 hi = *(const half8*)(p + 16);
    v16h r;
#pragma unroll
    for (int e = 0; e < 8; ++e) { r[e] = lo[e]; r[8 + e] = hi[e]; }
    return r;
}

// 16-bit B 32x16: lane%16 = N column, elems 0..7 -> K = 8*lh + 0..7,
//                 elems 8..15 -> K = 16 + 8*lh + 0..7
__device__ __forceinline__ v16h b_frag_from_lds(const _Float16* p) {
    half8 lo = *(const half8*)(p);
    half8 hi = *(const half8*)(p + 16);
    v16h r;
#pragma unroll
    for (int e = 0; e < 8; ++e) { r[e] = lo[e]; r[8 + e] = hi[e]; }
    return r;
}

// f32 -> f16 weight staging into LDS (vectorized 8 elems / iter)
template <int ROWS, int COLS, int LD>
__device__ __forceinline__ void stage_w(_Float16* dst, const float* src, int tid) {
    constexpr int G = ROWS * COLS / 8;
    for (int g = tid; g < G; g += BLOCK_THREADS) {
        int r = g / (COLS / 8);
        int c = (g % (COLS / 8)) * 8;
        fx4 x0 = *(const fx4*)(src + r * COLS + c);
        fx4 x1 = *(const fx4*)(src + r * COLS + c + 4);
        half8 h;
#pragma unroll
        for (int e = 0; e < 4; ++e) { h[e] = (_Float16)x0[e]; h[4 + e] = (_Float16)x1[e]; }
        *(half8*)(dst + r * LD + c) = h;
    }
}

__global__ void __launch_bounds__(BLOCK_THREADS)
fraud_hybrid_fused(const float* __restrict__ X,  const float* __restrict__ W1,
                   const float* __restrict__ b1, const float* __restrict__ W2,
                   const float* __restrict__ b2, const float* __restrict__ W3,
                   const float* __restrict__ b3, const float* __restrict__ rot,
                   const float* __restrict__ ent, const float* __restrict__ Wc,
                   const float* __restrict__ bc,  float* __restrict__ out) {
    extern __shared__ char smem[];
    _Float16* sW1  = (_Float16*)(smem + OFF_W1);
    _Float16* sH1  = (_Float16*)(smem + OFF_H1);
    _Float16* sW2  = (_Float16*)(smem + OFF_W2);
    _Float16* sW3  = (_Float16*)(smem + OFF_W3);
    _Float16* sH2  = (_Float16*)(smem + OFF_H2);
    float*    sLat = (float*)   (smem + OFF_LAT);
    float*    sFcl = (float*)   (smem + OFF_FCL);

    const int tid  = threadIdx.x;
    const int wave = tid >> 5;
    const int lane = tid & 31;
    const int l15  = lane & 15;
    const int lh   = lane >> 4;   // 0 or 1

    stage_w<H1_DIM,  INPUT_DIM, LDW1>(sW1, W1, tid);
    stage_w<H2_DIM,  H1_DIM,    LDW2>(sW2, W2, tid);
    stage_w<LAT_DIM, H2_DIM,    LDW3>(sW3, W3, tid);
    if (tid == 0) {   // block-wide constant: fcl = mean(cos(rotation))
        float fs = 0.f;
        for (int p = 0; p < 96; ++p) fs += __cosf(rot[p]);
        sFcl[0] = fs * (1.f / 96.f);
    }
    __syncthreads();

    const int rowBase = blockIdx.x * ROWS_PER_BLOCK + wave * 16;

    // -------- layer 1: h1 = relu(X @ W1^T + b1), 16x128 tile, K=512 --------
    v8f acc1[8];
#pragma unroll
    for (int nt = 0; nt < 8; ++nt)
#pragma unroll
        for (int e = 0; e < 8; ++e) acc1[nt][e] = 0.f;

    const float* xrow = X + (size_t)(rowBase + l15) * INPUT_DIM + lh * 8;
#pragma unroll 4
    for (int kk = 0; kk < 16; ++kk) {
        __builtin_prefetch(xrow + (kk + 4) * 32, 0, 3);   // WGP-scope global_prefetch_b8
        v16h a = a_frag_from_global(xrow + kk * 32);
#pragma unroll
        for (int nt = 0; nt < 8; ++nt) {
            v16h b = b_frag_from_lds(sW1 + (nt * 16 + l15) * LDW1 + kk * 32 + lh * 8);
            acc1[nt] = __builtin_amdgcn_wmma_f32_16x16x32_f16(
                false, a, false, b, (short)0, acc1[nt], false, false);
        }
    }
#pragma unroll
    for (int nt = 0; nt < 8; ++nt) {
        float bias = b1[nt * 16 + l15];
#pragma unroll
        for (int i = 0; i < 8; ++i) {
            float x = acc1[nt][i] + bias;
            x = x > 0.f ? x : 0.f;
            sH1[(wave * 16 + i + 8 * lh) * LDH1 + nt * 16 + l15] = (_Float16)x;
        }
    }
    __syncthreads();

    // -------- layer 2: h2 = relu(h1 @ W2^T + b2), 16x64 tile, K=128 --------
    v8f acc2[4];
#pragma unroll
    for (int nt = 0; nt < 4; ++nt)
#pragma unroll
        for (int e = 0; e < 8; ++e) acc2[nt][e] = 0.f;

    const _Float16* h1row = sH1 + (wave * 16 + l15) * LDH1 + lh * 8;
#pragma unroll
    for (int kk = 0; kk < 4; ++kk) {
        v16h a = a_frag_from_lds(h1row + kk * 32);
#pragma unroll
        for (int nt = 0; nt < 4; ++nt) {
            v16h b = b_frag_from_lds(sW2 + (nt * 16 + l15) * LDW2 + kk * 32 + lh * 8);
            acc2[nt] = __builtin_amdgcn_wmma_f32_16x16x32_f16(
                false, a, false, b, (short)0, acc2[nt], false, false);
        }
    }
#pragma unroll
    for (int nt = 0; nt < 4; ++nt) {
        float bias = b2[nt * 16 + l15];
#pragma unroll
        for (int i = 0; i < 8; ++i) {
            float x = acc2[nt][i] + bias;
            x = x > 0.f ? x : 0.f;
            sH2[(wave * 16 + i + 8 * lh) * LDH2 + nt * 16 + l15] = (_Float16)x;
        }
    }
    __syncthreads();

    // -------- layer 3: latent = h2 @ W3^T + b3, 16x32 tile, K=64 ----------
    v8f acc3[2];
#pragma unroll
    for (int nt = 0; nt < 2; ++nt)
#pragma unroll
        for (int e = 0; e < 8; ++e) acc3[nt][e] = 0.f;

    const _Float16* h2row = sH2 + (wave * 16 + l15) * LDH2 + lh * 8;
#pragma unroll
    for (int kk = 0; kk < 2; ++kk) {
        v16h a = a_frag_from_lds(h2row + kk * 32);
#pragma unroll
        for (int nt = 0; nt < 2; ++nt) {
            v16h b = b_frag_from_lds(sW3 + (nt * 16 + l15) * LDW3 + kk * 32 + lh * 8);
            acc3[nt] = __builtin_amdgcn_wmma_f32_16x16x32_f16(
                false, a, false, b, (short)0, acc3[nt], false, false);
        }
    }
#pragma unroll
    for (int nt = 0; nt < 2; ++nt) {
        float bias = b3[nt * 16 + l15];
#pragma unroll
        for (int i = 0; i < 8; ++i)
            sLat[(wave * 16 + i + 8 * lh) * LDLAT + nt * 16 + l15] = acc3[nt][i] + bias;
    }
    __syncthreads();

    // -------- per-row attention + entangle + classifier (f32 VALU) --------
    if (tid < ROWS_PER_BLOCK) {
        float* latp = sLat + tid * LDLAT;

        float kv[32], vv[32];
#pragma unroll
        for (int j = 0; j < 32; ++j) {
            float lj = latp[j];
            kv[j] = tanhf(lj * rot[32 + j]);
            vv[j] = lj * rot[64 + j];
        }
        const float inv_sqrt_d = 0.17677669529663687f; // 1/sqrt(32)

        // exp argument |q*k| < 1/sqrt(32) -> plain exp-sum == softmax exactly
        for (int i = 0; i < 32; ++i) {
            float q = tanhf(latp[i] * rot[i]) * inv_sqrt_d;
            float sum = 0.f, accv = 0.f;
#pragma unroll
            for (int j = 0; j < 32; ++j) {
                float e = __expf(q * kv[j]);
                sum  += e;
                accv += e * vv[j];
            }
            latp[i] = accv / sum;   // overwrite latent with attn (after last read of lat[i])
        }

        const float fcl = sFcl[0];
        float logit = bc[0];
        for (int i = 0; i < 32; ++i) {
            float a2 = latp[i] + (i < 31 ? ent[i] * latp[i + 1] : 0.f);
            logit += (a2 + fcl) * Wc[i];
        }
        out[blockIdx.x * ROWS_PER_BLOCK + tid] = 1.f / (1.f + __expf(-logit));
    }
}

extern "C" void kernel_launch(void* const* d_in, const int* in_sizes, int n_in,
                              void* d_out, int out_size, void* d_ws, size_t ws_size,
                              hipStream_t stream) {
    (void)n_in; (void)out_size; (void)d_ws; (void)ws_size;
    const float* X   = (const float*)d_in[0];
    const float* W1  = (const float*)d_in[1];
    const float* b1  = (const float*)d_in[2];
    const float* W2  = (const float*)d_in[3];
    const float* b2  = (const float*)d_in[4];
    const float* W3  = (const float*)d_in[5];
    const float* b3  = (const float*)d_in[6];
    const float* rot = (const float*)d_in[7];
    const float* ent = (const float*)d_in[8];
    const float* Wc  = (const float*)d_in[9];
    const float* bc  = (const float*)d_in[10];
    float* out = (float*)d_out;

    const int batch = in_sizes[0] / INPUT_DIM;       // 131072
    const int grid  = batch / ROWS_PER_BLOCK;        // 1024

    (void)hipFuncSetAttribute(reinterpret_cast<const void*>(fraud_hybrid_fused),
                              hipFuncAttributeMaxDynamicSharedMemorySize, SMEM_BYTES);
    fraud_hybrid_fused<<<grid, BLOCK_THREADS, SMEM_BYTES, stream>>>(
        X, W1, b1, W2, b2, W3, b3, rot, ent, Wc, bc, out);
}
